// CausalSelfAttention_9629316678114
// MI455X (gfx1250) — compile-verified, matrix operand order checked by reading, not run on hardware
//
#include <hip/hip_runtime.h>
#include <hip/hip_bf16.h>

// ---------------------------------------------------------------------------
// CausalSelfAttention forward for MI455X (gfx1250, wave32, WMMA)
// B=2, T=2048, C=1024, H=16, Dh=64
// ---------------------------------------------------------------------------

typedef __attribute__((ext_vector_type(16))) _Float16 v16h;
typedef __attribute__((ext_vector_type(8)))  _Float16 v8h;
typedef __attribute__((ext_vector_type(8)))  float    v8f;

#define B_SZ   2
#define T_SEQ  2048
#define C_DIM  1024
#define H_NUM  16
#define DH     64

__device__ __forceinline__ v8f wmma_f16(v16h a, v16h b, v8f c) {
  return __builtin_amdgcn_wmma_f32_16x16x32_f16(
      /*neg_a=*/false, a, /*neg_b=*/false, b,
      /*c_mod=*/(short)0, c, /*reuse_a=*/false, /*reuse_b=*/false);
}

// Load a 16x32 f16 A-fragment (or a B-fragment whose memory holds B^T
// row-major: "row n contains the K elements") from row-major memory.
// ISA 7.12.2 16-bit A 16x32 layout.
__device__ __forceinline__ v16h load_frag_rm(const _Float16* p, int ld) {
  int lane = threadIdx.x & 31;
  int row  = lane & 15;
  int hi   = (lane >> 4) ? 8 : 0;
  const _Float16* r = p + (size_t)row * ld;
  v8h lo = *(const v8h*)(r + hi);
  v8h hb = *(const v8h*)(r + 16 + hi);
  v16h out;
#pragma unroll
  for (int i = 0; i < 8; ++i) { out[i] = lo[i]; out[i + 8] = hb[i]; }
  return out;
}

// Gather a 32x16 B-fragment (rows = contraction K, cols = N) from an LDS
// tile stored row-major [32][ldh] (rows = keys, cols = Dh).
__device__ __forceinline__ v16h load_b_frag_lds(const _Float16* vl, int ldh, int n0) {
  int lane = threadIdx.x & 31;
  int n  = lane & 15;
  int hi = (lane >> 4) ? 8 : 0;
  v16h out;
#pragma unroll
  for (int j = 0; j < 16; ++j) {
    int k = ((j >> 3) * 16) + hi + (j & 7);
    out[j] = vl[k * ldh + n0 + n];
  }
  return out;
}

// ---------------------------------------------------------------------------
// f32 -> f16 cast
// ---------------------------------------------------------------------------
__global__ void cast_f32_f16(const float* __restrict__ in,
                             _Float16* __restrict__ out, int n) {
  int i = blockIdx.x * 256 + threadIdx.x;
  if (i < n) out[i] = (_Float16)in[i];
}

// ---------------------------------------------------------------------------
// QKV GEMM: qkv[m,n] = sum_k x[m,k] * W_attn[n,k] + b_attn[n]
// M=4096, N=3072, K=1024.
// Block = 256 threads = 8 waves arranged 2(M) x 4(N); wave tile = 32 x 64.
// K unrolled by 2 with ping-pong register sets: loads for step i+1 are
// issued before the WMMAs of step i, and each load lands directly in the
// set consumed next -> no rotation copies.
// Epilogue scatters into Q/K/V f16 buffers, layout [B,H,T,Dh].
// ---------------------------------------------------------------------------
__global__ void __launch_bounds__(256)
gemm_qkv(const _Float16* __restrict__ xh, const _Float16* __restrict__ wh,
         const float* __restrict__ bias,
         _Float16* __restrict__ qb, _Float16* __restrict__ kb,
         _Float16* __restrict__ vb) {
  int wave = threadIdx.x >> 5;        // 0..7
  int wm = wave >> 2;                 // 0..1
  int wn = wave & 3;                  // 0..3
  int m0 = blockIdx.y * 64 + wm * 32;
  int n0 = blockIdx.x * 256 + wn * 64;

  const _Float16* Arow0 = xh + (size_t)m0 * C_DIM;
  const _Float16* Arow1 = xh + (size_t)(m0 + 16) * C_DIM;
  const _Float16* Brow[4];
#pragma unroll
  for (int t = 0; t < 4; ++t) Brow[t] = wh + (size_t)(n0 + t * 16) * C_DIM;

  v8f acc[2][4] = {};

  // prologue: set0 <- k = 0
  v16h a0A = load_frag_rm(Arow0, C_DIM);
  v16h a1A = load_frag_rm(Arow1, C_DIM);
  v16h bA[4];
#pragma unroll
  for (int t = 0; t < 4; ++t) bA[t] = load_frag_rm(Brow[t], C_DIM);

  v16h a0B, a1B, bB[4];

  for (int k = 0; k < C_DIM; k += 64) {
    // load set1 <- k+32 (always in range: k <= K-64)
    a0B = load_frag_rm(Arow0 + k + 32, C_DIM);
    a1B = load_frag_rm(Arow1 + k + 32, C_DIM);
#pragma unroll
    for (int t = 0; t < 4; ++t) bB[t] = load_frag_rm(Brow[t] + k + 32, C_DIM);

    // WMMAs on set0 (k)
#pragma unroll
    for (int t = 0; t < 4; ++t) {
      acc[0][t] = wmma_f16(a0A, bA[t], acc[0][t]);
      acc[1][t] = wmma_f16(a1A, bA[t], acc[1][t]);
    }

    // load set0 <- k+64 (clamped dummy on final iteration; never consumed)
    int kn = (k + 64 < C_DIM) ? (k + 64) : 0;
    a0A = load_frag_rm(Arow0 + kn, C_DIM);
    a1A = load_frag_rm(Arow1 + kn, C_DIM);
#pragma unroll
    for (int t = 0; t < 4; ++t) bA[t] = load_frag_rm(Brow[t] + kn, C_DIM);

    // WMMAs on set1 (k+32)
#pragma unroll
    for (int t = 0; t < 4; ++t) {
      acc[0][t] = wmma_f16(a0B, bB[t], acc[0][t]);
      acc[1][t] = wmma_f16(a1B, bB[t], acc[1][t]);
    }
  }

  int lane = threadIdx.x & 31;
  int ncl  = lane & 15;
  int mhi  = (lane >> 4) ? 8 : 0;
#pragma unroll
  for (int t = 0; t < 4; ++t) {
    int n = n0 + t * 16 + ncl;
    float bv = bias[n];
    int which = n >> 10;            // 0=q,1=k,2=v
    int c = n & 1023;
    int h = c >> 6;
    int d = c & 63;
    _Float16* dst = (which == 0) ? qb : ((which == 1) ? kb : vb);
#pragma unroll
    for (int mi = 0; mi < 2; ++mi) {
#pragma unroll
      for (int r = 0; r < 8; ++r) {
        int m = m0 + mi * 16 + r + mhi;
        int bidx = m >> 11;           // /T
        int tpos = m & (T_SEQ - 1);
        dst[(((size_t)(bidx * H_NUM + h) * T_SEQ) + tpos) * DH + d] =
            (_Float16)(acc[mi][t][r] + bv);
      }
    }
  }
}

// ---------------------------------------------------------------------------
// Flash attention with ALiBi + causal mask.
// Block = 4 waves; wave handles 16 q-rows; grid = (T/64, H, B).
// ---------------------------------------------------------------------------
#define P_LD   40                 // 16 rows * 40 halves (80B rows, 16B aligned)
#define V_LD   72                 // 32 rows * 72 halves (144B rows, 16B aligned)
#define WAVE_LDS (16 * P_LD + 32 * V_LD)   // 640 + 2304 = 2944 halves

__global__ void __launch_bounds__(128)
attn_fwd(const _Float16* __restrict__ qbuf, const _Float16* __restrict__ kbuf,
         const _Float16* __restrict__ vbuf, _Float16* __restrict__ ybuf) {
  __shared__ _Float16 smem[4 * WAVE_LDS];

  int wave = threadIdx.x >> 5;
  int lane = threadIdx.x & 31;
  int b = blockIdx.z;
  int h = blockIdx.y;
  int q0 = blockIdx.x * 64 + wave * 16;

  const _Float16* Q = qbuf + (size_t)(b * H_NUM + h) * T_SEQ * DH;
  const _Float16* K = kbuf + (size_t)(b * H_NUM + h) * T_SEQ * DH;
  const _Float16* V = vbuf + (size_t)(b * H_NUM + h) * T_SEQ * DH;

  _Float16* pl = smem + wave * WAVE_LDS;        // P tile  16 x P_LD
  _Float16* vl = pl + 16 * P_LD;                // V tile  32 x V_LD

  // Q fragments held for the whole key loop (Dh=64 -> two 16x32 fragments)
  v16h aq0 = load_frag_rm(Q + (size_t)q0 * DH + 0, DH);
  v16h aq1 = load_frag_rm(Q + (size_t)q0 * DH + 32, DH);

  v8f o[4] = {};
  float mrow[8], lrow[8];
#pragma unroll
  for (int r = 0; r < 8; ++r) { mrow[r] = -1e30f; lrow[r] = 0.0f; }

  float slope = exp2f(-8.0f / (float)H_NUM * (float)(h + 1));
  int mhi = (lane >> 4) ? 8 : 0;
  int ncl = lane & 15;

  for (int kb0 = 0; kb0 <= q0 + 15; kb0 += 32) {
    // ---- stage V[kb0..kb0+31, 0..63] into LDS (wave-cooperative) ----
#pragma unroll
    for (int c = 0; c < 8; ++c) {
      int idx = c * 32 + lane;        // 256 chunks of 8 halves
      int rr = idx >> 3;
      int ch = idx & 7;
      v8h d = *(const v8h*)(V + (size_t)(kb0 + rr) * DH + ch * 8);
      *(v8h*)(vl + rr * V_LD + ch * 8) = d;
    }

    // ---- S = Q K^T for two 16-key tiles ----
    v8f s[2];
#pragma unroll
    for (int t2 = 0; t2 < 2; ++t2) {
      v16h bk0 = load_frag_rm(K + (size_t)(kb0 + t2 * 16) * DH + 0, DH);
      v16h bk1 = load_frag_rm(K + (size_t)(kb0 + t2 * 16) * DH + 32, DH);
      v8f c = {};
      c = wmma_f16(aq0, bk0, c);
      c = wmma_f16(aq1, bk1, c);
      s[t2] = c;
    }

    // ---- scale + ALiBi + causal mask + online softmax ----
    float mnew[8];
#pragma unroll
    for (int r = 0; r < 8; ++r) {
      int qr = q0 + r + mhi;
      float mx = mrow[r];
#pragma unroll
      for (int t2 = 0; t2 < 2; ++t2) {
        int kc = kb0 + t2 * 16 + ncl;
        float v = s[t2][r] * 0.125f +
                  slope * fminf((float)(kc - qr), 0.0f);
        if (kc > qr) v = -1e30f;
        s[t2][r] = v;
        mx = fmaxf(mx, v);
      }
#pragma unroll
      for (int off = 1; off < 16; off <<= 1)
        mx = fmaxf(mx, __shfl_xor(mx, off, 16));
      mnew[r] = mx;
    }

#pragma unroll
    for (int r = 0; r < 8; ++r) {
      float alpha = __expf(mrow[r] - mnew[r]);
      float rowsum = 0.0f;
#pragma unroll
      for (int t2 = 0; t2 < 2; ++t2) {
        float p = __expf(s[t2][r] - mnew[r]);
        s[t2][r] = p;
        rowsum += p;
      }
#pragma unroll
      for (int off = 1; off < 16; off <<= 1)
        rowsum += __shfl_xor(rowsum, off, 16);
      lrow[r] = lrow[r] * alpha + rowsum;
      mrow[r] = mnew[r];
#pragma unroll
      for (int d4 = 0; d4 < 4; ++d4) o[d4][r] *= alpha;
    }

    // ---- C-layout P -> LDS -> A-layout fragment (same-wave transpose) ----
#pragma unroll
    for (int t2 = 0; t2 < 2; ++t2)
#pragma unroll
      for (int r = 0; r < 8; ++r)
        pl[(r + mhi) * P_LD + t2 * 16 + ncl] = (_Float16)s[t2][r];

    // Same-wave LDS ops are in-order; make the dependency explicit.
    asm volatile("s_wait_dscnt 0" ::: "memory");

    v16h ap = load_frag_rm(pl, P_LD);

    // ---- O += P * V ----
#pragma unroll
    for (int d4 = 0; d4 < 4; ++d4) {
      v16h bv = load_b_frag_lds(vl, V_LD, d4 * 16);
      o[d4] = wmma_f16(ap, bv, o[d4]);
    }
  }

  // ---- epilogue: normalize and store y as f16, layout [B,T,C] ----
#pragma unroll
  for (int d4 = 0; d4 < 4; ++d4) {
#pragma unroll
    for (int r = 0; r < 8; ++r) {
      int t = q0 + r + mhi;
      float val = o[d4][r] / lrow[r];
      ybuf[((size_t)(b * T_SEQ + t)) * C_DIM + h * DH + d4 * 16 + ncl] =
          (_Float16)val;
    }
  }
}

// ---------------------------------------------------------------------------
// Output projection: out[m,n] = sum_k y[m,k] * W_proj[n,k] + b_proj[n]  (f32)
// M=4096, N=1024, K=1024.  Same ping-pong pipelined 32x64-per-wave structure.
// ---------------------------------------------------------------------------
__global__ void __launch_bounds__(256)
gemm_proj(const _Float16* __restrict__ yh, const _Float16* __restrict__ wh,
          const float* __restrict__ bias, float* __restrict__ out) {
  int wave = threadIdx.x >> 5;
  int wm = wave >> 2;
  int wn = wave & 3;
  int m0 = blockIdx.y * 64 + wm * 32;
  int n0 = blockIdx.x * 256 + wn * 64;

  const _Float16* Arow0 = yh + (size_t)m0 * C_DIM;
  const _Float16* Arow1 = yh + (size_t)(m0 + 16) * C_DIM;
  const _Float16* Brow[4];
#pragma unroll
  for (int t = 0; t < 4; ++t) Brow[t] = wh + (size_t)(n0 + t * 16) * C_DIM;

  v8f acc[2][4] = {};

  v16h a0A = load_frag_rm(Arow0, C_DIM);
  v16h a1A = load_frag_rm(Arow1, C_DIM);
  v16h bA[4];
#pragma unroll
  for (int t = 0; t < 4; ++t) bA[t] = load_frag_rm(Brow[t], C_DIM);

  v16h a0B, a1B, bB[4];

  for (int k = 0; k < C_DIM; k += 64) {
    a0B = load_frag_rm(Arow0 + k + 32, C_DIM);
    a1B = load_frag_rm(Arow1 + k + 32, C_DIM);
#pragma unroll
    for (int t = 0; t < 4; ++t) bB[t] = load_frag_rm(Brow[t] + k + 32, C_DIM);

#pragma unroll
    for (int t = 0; t < 4; ++t) {
      acc[0][t] = wmma_f16(a0A, bA[t], acc[0][t]);
      acc[1][t] = wmma_f16(a1A, bA[t], acc[1][t]);
    }

    int kn = (k + 64 < C_DIM) ? (k + 64) : 0;
    a0A = load_frag_rm(Arow0 + kn, C_DIM);
    a1A = load_frag_rm(Arow1 + kn, C_DIM);
#pragma unroll
    for (int t = 0; t < 4; ++t) bA[t] = load_frag_rm(Brow[t] + kn, C_DIM);

#pragma unroll
    for (int t = 0; t < 4; ++t) {
      acc[0][t] = wmma_f16(a0B, bB[t], acc[0][t]);
      acc[1][t] = wmma_f16(a1B, bB[t], acc[1][t]);
    }
  }

  int lane = threadIdx.x & 31;
  int ncl  = lane & 15;
  int mhi  = (lane >> 4) ? 8 : 0;
#pragma unroll
  for (int t = 0; t < 4; ++t) {
    int n = n0 + t * 16 + ncl;
    float bv = bias[n];
#pragma unroll
    for (int mi = 0; mi < 2; ++mi) {
#pragma unroll
      for (int r = 0; r < 8; ++r) {
        int m = m0 + mi * 16 + r + mhi;
        out[(size_t)m * C_DIM + n] = acc[mi][t][r] + bv;
      }
    }
  }
}

// ---------------------------------------------------------------------------
// Launch
// ---------------------------------------------------------------------------
extern "C" void kernel_launch(void* const* d_in, const int* in_sizes, int n_in,
                              void* d_out, int out_size, void* d_ws, size_t ws_size,
                              hipStream_t stream) {
  const float* x      = (const float*)d_in[0];   // [B,T,C]
  const float* W_attn = (const float*)d_in[1];   // [3C,C]
  const float* b_attn = (const float*)d_in[2];   // [3C]
  const float* W_proj = (const float*)d_in[3];   // [C,C]
  const float* b_proj = (const float*)d_in[4];   // [C]
  float* out = (float*)d_out;

  char* ws = (char*)d_ws;
  const size_t n_x  = (size_t)B_SZ * T_SEQ * C_DIM;        // 4,194,304
  const size_t n_wa = (size_t)3 * C_DIM * C_DIM;           // 3,145,728
  const size_t n_wp = (size_t)C_DIM * C_DIM;               // 1,048,576
  const size_t n_qkv = (size_t)B_SZ * H_NUM * T_SEQ * DH;  // 4,194,304

  _Float16* xh  = (_Float16*)(ws);
  _Float16* wah = (_Float16*)(ws + 2 * n_x);
  _Float16* wph = (_Float16*)(ws + 2 * (n_x + n_wa));
  _Float16* qb  = (_Float16*)(ws + 2 * (n_x + n_wa + n_wp));
  _Float16* kb  = (_Float16*)(ws + 2 * (n_x + n_wa + n_wp + n_qkv));
  _Float16* vb  = (_Float16*)(ws + 2 * (n_x + n_wa + n_wp + 2 * n_qkv));
  _Float16* yb  = (_Float16*)(ws + 2 * (n_x + n_wa + n_wp + 3 * n_qkv));

  cast_f32_f16<<<(int)((n_x  + 255) / 256), 256, 0, stream>>>(x, xh, (int)n_x);
  cast_f32_f16<<<(int)((n_wa + 255) / 256), 256, 0, stream>>>(W_attn, wah, (int)n_wa);
  cast_f32_f16<<<(int)((n_wp + 255) / 256), 256, 0, stream>>>(W_proj, wph, (int)n_wp);

  // QKV GEMM: block tile 64x256 -> grid = (3072/256, 4096/64) = (12, 64)
  gemm_qkv<<<dim3(12, 64), 256, 0, stream>>>(xh, wah, b_attn, qb, kb, vb);

  // Attention: (T/64, H, B)
  attn_fwd<<<dim3(T_SEQ / 64, H_NUM, B_SZ), 128, 0, stream>>>(qb, kb, vb, yb);

  // Projection: block tile 64x256 -> grid = (1024/256, 4096/64) = (4, 64)
  gemm_proj<<<dim3(4, 64), 256, 0, stream>>>(yb, wph, b_proj, out);
}
